// DeltaNetAttention_34531537059993
// MI455X (gfx1250) — matrix operand order repaired and checked by reading, not compiled, not run on hardware
//
#include <hip/hip_runtime.h>
#include <hip/hip_bf16.h>

// Problem constants (reference: B=8, T=4096, D=512, H=4, C=64, d=128, dv=256)
#define BB   8
#define TT   4096
#define DD   512
#define HH   4
#define CC   64
#define HD   128
#define DV   256
#define NC   (TT / CC)          // 64 chunks
#define MTOK (BB * TT)          // 32768 token rows

typedef __attribute__((ext_vector_type(16))) _Float16 v16h;
typedef __attribute__((ext_vector_type(8)))  float    v8f;

#if __has_builtin(__builtin_amdgcn_global_load_async_to_lds_b128) && \
    __has_builtin(__builtin_amdgcn_s_wait_asynccnt)
#define USE_ASYNC_LDS 1
// Pointer types required by the async-copy builtin (from hipcc diagnostic:
// arg0 is 'int __vector(4) __device__*' i.e. AS1; arg1 is the LDS-side ptr).
typedef int v4i __attribute__((vector_size(16)));
typedef __attribute__((address_space(1))) v4i v4i_glob;
typedef __attribute__((address_space(3))) v4i v4i_lds;
#else
#define USE_ASYNC_LDS 0
#endif

// ---------------------------------------------------------------------------
// WMMA fragment loaders (wave32; this code's k order is contiguous, k = e + 16*hi, in A and B)
// ---------------------------------------------------------------------------

// A-matrix 16x32 f16, row-major source with row stride lda (halves).
// lane m = lane&15, hi = lane>=16. K(e) = e + 16*hi.
__device__ inline v16h load_frag_a(const _Float16* p0, int lda, int lane) {
  const int m = lane & 15, hi = (lane >> 4) & 1;
  const _Float16* p = p0 + (size_t)m * lda + 16 * hi;
  v16h a;
#pragma unroll
  for (int e = 0; e < 16; ++e) a[e] = p[e];
  return a;
}

// B-matrix 32x16 f16 where logical B[k][n] = srcT[n][k] (pre-transposed,
// row stride ldt). lane n = lane&15; K(e) = e + 16*hi -> contiguous loads.
__device__ inline v16h load_frag_bT(const _Float16* p0, int ldt, int lane) {
  const int n = lane & 15, hi = (lane >> 4) & 1;
  const _Float16* p = p0 + (size_t)n * ldt + 16 * hi;
  v16h b;
#pragma unroll
  for (int e = 0; e < 16; ++e) b[e] = p[e];
  return b;
}

// B-matrix 32x16 f16 from row-major B[k][n], row stride ldb (strided gather,
// used only against LDS-resident tiles).
__device__ inline v16h load_frag_b(const _Float16* p0, int ldb, int lane) {
  const int n = lane & 15, hi = (lane >> 4) & 1;
  const _Float16* p = p0 + n + (size_t)(16 * hi) * ldb;
  v16h b;
#pragma unroll
  for (int e = 0; e < 16; ++e) b[e] = p[(size_t)e * ldb];
  return b;
}

// ---------------------------------------------------------------------------
// Generic f16 GEMM:  C[M,N] = A[M,K] @ W[K,N], W given transposed (Wt[N,K]).
// block = (32,4): 4 waves stack 4 row-tiles; grid = (N/16, M/64).
// Writes f32 (Cf) or f16 (Ch).
// ---------------------------------------------------------------------------
__global__ __launch_bounds__(128) void gemm_wmma_kernel(
    const _Float16* __restrict__ A, const _Float16* __restrict__ Wt,
    float* __restrict__ Cf, _Float16* __restrict__ Ch, int N, int K) {
  const int lane = threadIdx.x & 31;
  const int m0 = (blockIdx.y * 4 + threadIdx.y) * 16;
  const int n0 = blockIdx.x * 16;
  const _Float16* arow = A + (size_t)m0 * K;
  const _Float16* brow = Wt + (size_t)n0 * K;
  v8f acc = {};
  for (int k0 = 0; k0 < K; k0 += 32) {
    if (k0 + 64 < K) {  // speculative prefetch of next-next k-slab
      __builtin_prefetch(arow + k0 + 64, 0, 1);
      __builtin_prefetch(brow + k0 + 64, 0, 1);
    }
    v16h a = load_frag_a(arow + k0, K, lane);
    v16h b = load_frag_bT(brow + k0, K, lane);
    acc = __builtin_amdgcn_wmma_f32_16x16x32_f16(false, a, false, b,
                                                 (short)0, acc, false, false);
  }
  const int n = lane & 15, hi = (lane >> 4) & 1;
#pragma unroll
  for (int r = 0; r < 8; ++r) {
    const int m = m0 + r + 8 * hi;
    const size_t o = (size_t)m * N + n0 + n;
    if (Cf) Cf[o] = acc[r];
    else    Ch[o] = (_Float16)acc[r];
  }
}

// ---------------------------------------------------------------------------
// Per-(token,head) RMSNorm over head_dim=128, in-place on f16 buffer.
// ---------------------------------------------------------------------------
__global__ __launch_bounds__(128) void rmsnorm_f16_kernel(_Float16* __restrict__ q) {
  __shared__ float red[128];
  const size_t base = (size_t)blockIdx.x * HD;
  const int tid = threadIdx.x;
  const float v = (float)q[base + tid];
  red[tid] = v * v;
  __syncthreads();
  for (int s = 64; s > 0; s >>= 1) {
    if (tid < s) red[tid] += red[tid + s];
    __syncthreads();
  }
  const float scale = rsqrtf(red[0] * (1.0f / HD) + 1.1920929e-07f);
  q[base + tid] = (_Float16)(v * scale);
}

// beta = sigmoid(x @ Wbeta): one thread per token computes all H=4 heads
// (Wbeta row is a 16B broadcast; x row read once per 4 outputs).
__global__ __launch_bounds__(256) void beta_kernel(const float* __restrict__ x,
                                                   const float* __restrict__ Wb,
                                                   float* __restrict__ beta) {
  const int tok = blockIdx.x * blockDim.x + threadIdx.x;
  if (tok >= MTOK) return;
  const float* xr = x + (size_t)tok * DD;
  float s0 = 0.f, s1 = 0.f, s2 = 0.f, s3 = 0.f;
  for (int i = 0; i < DD; ++i) {
    const float xv = xr[i];
    const float* wr = Wb + i * HH;
    s0 = fmaf(xv, wr[0], s0);
    s1 = fmaf(xv, wr[1], s1);
    s2 = fmaf(xv, wr[2], s2);
    s3 = fmaf(xv, wr[3], s3);
  }
  float* br = beta + (size_t)tok * HH;
  br[0] = 1.0f / (1.0f + __expf(-s0));
  br[1] = 1.0f / (1.0f + __expf(-s1));
  br[2] = 1.0f / (1.0f + __expf(-s2));
  br[3] = 1.0f / (1.0f + __expf(-s3));
}

__global__ void cvt_f32_f16_kernel(const float* __restrict__ src,
                                   _Float16* __restrict__ dst, int n) {
  const int i = blockIdx.x * blockDim.x + threadIdx.x;
  if (i < n) dst[i] = (_Float16)src[i];
}

// Transpose + convert weight W[K,N] f32 -> Wt[N,K] f16 (coalesced writes).
__global__ void cvt_transpose_kernel(const float* __restrict__ W,
                                     _Float16* __restrict__ Wt, int K, int N) {
  const int idx = blockIdx.x * blockDim.x + threadIdx.x;  // = n*K + k
  if (idx >= K * N) return;
  const int k = idx % K, n = idx / K;
  Wt[idx] = (_Float16)W[(size_t)k * N + n];
}

// ---------------------------------------------------------------------------
// Chunked delta-scan:  per (b, chunk, head) block of 256 threads (8 waves):
//   W[t][s] = b_s * prod_{r=s+1..t}(1-b_r)  (lower-triangular decay mask)
//   A = Q K^T  (64x64, K=128)     -> WMMA
//   O = (A .* W) V (64x256, K=64) -> WMMA (V staged in LDS via async copy)
// ---------------------------------------------------------------------------
#define SV_LD 264   // padded LDS row stride for the V tile (halves)

__global__ __launch_bounds__(256) void chunk_scan_kernel(
    const _Float16* __restrict__ qh, const _Float16* __restrict__ kh,
    const _Float16* __restrict__ vh, const float* __restrict__ beta,
    _Float16* __restrict__ oh) {
  __shared__ float    sBeta[CC];
  __shared__ float    sW[CC * CC];        // 16 KB
  __shared__ _Float16 sAW[CC * 72];       // padded stride 72
  __shared__ _Float16 sV[CC * SV_LD];     // 64 x 256 V tile, padded rows

  const int idx = blockIdx.x;             // ((b*NC)+c)*H + h
  const int h = idx & (HH - 1);
  const int c = (idx >> 2) & (NC - 1);
  const int b = idx >> 8;
  const int tid = threadIdx.x;
  const int wave = tid >> 5, lane = tid & 31;
  const int tokbase = b * TT + c * CC;
  const int n = lane & 15, hi = (lane >> 4) & 1;

  // ---- 0) kick off the V tile copy to LDS (ASYNCcnt path) first so the
  //         DMA overlaps the decay-matrix build and the Q K^T WMMAs.
  {
    const _Float16* vbase = vh + (size_t)tokbase * (HH * DV) + h * DV;
#pragma unroll
    for (int it = 0; it < 8; ++it) {
      const int f = tid + it * 256;       // 2048 x 16B packets
      const int row = f >> 5;             // 32 packets per 256-half row
      const int col = (f & 31) * 8;       // halves
      const _Float16* src = vbase + (size_t)row * (HH * DV) + col;
      _Float16* dst = &sV[row * SV_LD + col];
#if USE_ASYNC_LDS
      // AS1 side: global addresses == generic addresses (inttoptr).
      // AS3 side: LDS offset is the low 32 bits of the generic address.
      __builtin_amdgcn_global_load_async_to_lds_b128(
          (v4i_glob*)(uintptr_t)src,
          (v4i_lds*)(uint32_t)(uintptr_t)dst,
          0, 0);
#else
      *(uint4*)dst = *(const uint4*)src;
#endif
    }
  }

  // ---- 1) decay mask
  if (tid < CC) sBeta[tid] = beta[(size_t)(tokbase + tid) * HH + h];
  __syncthreads();
  if (tid < CC) {
    const int t = tid;
    float accd = 1.0f;
    for (int s = t; s >= 0; --s) {        // w[t][s] = b_s * prod_{r=s+1..t}(1-b_r)
      sW[t * CC + s] = sBeta[s] * accd;
      accd *= (1.0f - sBeta[s]);
    }
    for (int s = t + 1; s < CC; ++s) sW[t * CC + s] = 0.0f;
  }
  __syncthreads();

  // ---- 2) A = Q K^T, masked by W, stored f16 into LDS. 16 tiles, 2 per wave.
#pragma unroll
  for (int i = 0; i < 2; ++i) {
    const int tile = wave * 2 + i;
    const int ti = tile >> 2, si = tile & 3;
    const _Float16* qbase = qh + ((size_t)(tokbase + ti * 16) * HH + h) * HD;
    const _Float16* kbase = kh + ((size_t)(tokbase + si * 16) * HH + h) * HD;
    v8f acc = {};
#pragma unroll
    for (int k0 = 0; k0 < HD; k0 += 32) {
      v16h a  = load_frag_a (qbase + k0, HH * HD, lane);
      v16h bb = load_frag_bT(kbase + k0, HH * HD, lane);  // B[k][n]=K[s0+n][k0+k]
      acc = __builtin_amdgcn_wmma_f32_16x16x32_f16(false, a, false, bb,
                                                   (short)0, acc, false, false);
    }
#pragma unroll
    for (int r = 0; r < 8; ++r) {
      const int t = ti * 16 + r + 8 * hi;
      const int s = si * 16 + n;
      sAW[t * 72 + s] = (_Float16)(acc[r] * sW[t * CC + s]);
    }
  }

  // ---- 3) wait for this wave's async V packets, then block barrier so the
  //         full tile is visible to everyone.
#if USE_ASYNC_LDS
  __builtin_amdgcn_s_wait_asynccnt(0);
#endif
  __syncthreads();

  // ---- 4) O = AW @ V : 4x16 tiles of 16x16, 8 per wave, K=64 (2 wmma steps).
#pragma unroll
  for (int i = 0; i < 8; ++i) {
    const int tile = i * 8 + wave;
    const int tj = tile >> 4, nj = tile & 15;
    v8f acc = {};
#pragma unroll
    for (int k0 = 0; k0 < CC; k0 += 32) {
      v16h a  = load_frag_a(&sAW[(tj * 16) * 72 + k0], 72, lane);
      v16h bb = load_frag_b(&sV[k0 * SV_LD + nj * 16], SV_LD, lane);
      acc = __builtin_amdgcn_wmma_f32_16x16x32_f16(false, a, false, bb,
                                                   (short)0, acc, false, false);
    }
#pragma unroll
    for (int r = 0; r < 8; ++r) {
      const int t = tj * 16 + r + 8 * hi;
      oh[(size_t)(tokbase + t) * (HH * DV) + h * DV + nj * 16 + n] = (_Float16)acc[r];
    }
  }
}

// ---------------------------------------------------------------------------
// Launcher
// ---------------------------------------------------------------------------
extern "C" void kernel_launch(void* const* d_in, const int* in_sizes, int n_in,
                              void* d_out, int out_size, void* d_ws, size_t ws_size,
                              hipStream_t stream) {
  (void)in_sizes; (void)n_in; (void)out_size; (void)ws_size;
  const float* x  = (const float*)d_in[0];
  const float* Wq = (const float*)d_in[3];
  const float* Wk = (const float*)d_in[4];
  const float* Wv = (const float*)d_in[5];
  const float* Wb = (const float*)d_in[6];
  const float* Wp = (const float*)d_in[7];
  float* out = (float*)d_out;

  char* ws = (char*)d_ws;
  size_t off = 0;
  auto bump = [&](size_t bytes) -> void* {
    void* p = ws + off;
    off += (bytes + 255) & ~(size_t)255;
    return p;
  };
  _Float16* xh  = (_Float16*)bump((size_t)MTOK * DD * 2);        // 32 MiB
  _Float16* qh  = (_Float16*)bump((size_t)MTOK * DD * 2);        // 32 MiB
  _Float16* kh  = (_Float16*)bump((size_t)MTOK * DD * 2);        // 32 MiB
  _Float16* vh  = (_Float16*)bump((size_t)MTOK * HH * DV * 2);   // 64 MiB
  _Float16* oh  = (_Float16*)bump((size_t)MTOK * HH * DV * 2);   // 64 MiB
  float*    bet = (float*)   bump((size_t)MTOK * HH * 4);
  _Float16* WqT = (_Float16*)bump((size_t)DD * DD * 2);
  _Float16* WkT = (_Float16*)bump((size_t)DD * DD * 2);
  _Float16* WvT = (_Float16*)bump((size_t)DD * (HH * DV) * 2);
  _Float16* WpT = (_Float16*)bump((size_t)(HH * DV) * DD * 2);

  // 1) precision conversions
  {
    const int nx = MTOK * DD;
    cvt_f32_f16_kernel<<<(nx + 255) / 256, 256, 0, stream>>>(x, xh, nx);
  }
  cvt_transpose_kernel<<<(DD * DD + 255) / 256, 256, 0, stream>>>(Wq, WqT, DD, DD);
  cvt_transpose_kernel<<<(DD * DD + 255) / 256, 256, 0, stream>>>(Wk, WkT, DD, DD);
  cvt_transpose_kernel<<<(DD * HH * DV + 255) / 256, 256, 0, stream>>>(Wv, WvT, DD, HH * DV);
  cvt_transpose_kernel<<<(HH * DV * DD + 255) / 256, 256, 0, stream>>>(Wp, WpT, HH * DV, DD);

  // 2) projections (WMMA f16, f32 accumulate)
  const dim3 blk(32, 4);
  gemm_wmma_kernel<<<dim3(DD / 16, MTOK / 64), blk, 0, stream>>>(xh, WqT, nullptr, qh, DD, DD);
  gemm_wmma_kernel<<<dim3(DD / 16, MTOK / 64), blk, 0, stream>>>(xh, WkT, nullptr, kh, DD, DD);
  gemm_wmma_kernel<<<dim3((HH * DV) / 16, MTOK / 64), blk, 0, stream>>>(xh, WvT, nullptr, vh, HH * DV, DD);
  beta_kernel<<<(MTOK + 255) / 256, 256, 0, stream>>>(x, Wb, bet);

  // 3) per-head rmsnorm of q, k (in place)
  rmsnorm_f16_kernel<<<MTOK * HH, 128, 0, stream>>>(qh);
  rmsnorm_f16_kernel<<<MTOK * HH, 128, 0, stream>>>(kh);

  // 4) chunked delta scan (closed form per 64-token chunk) -> o
  chunk_scan_kernel<<<BB * NC * HH, 256, 0, stream>>>(qh, kh, vh, bet, oh);

  // 5) output projection -> f32 result
  gemm_wmma_kernel<<<dim3(DD / 16, MTOK / 64), blk, 0, stream>>>(oh, WpT, out, nullptr, DD, HH * DV);
}
